// LossUnsupervised_46119358825080
// MI455X (gfx1250) — matrix operand order repaired and checked by reading, not compiled, not run on hardware
//
#include <hip/hip_runtime.h>
#include <math.h>

typedef __attribute__((ext_vector_type(16))) _Float16 v16h;
typedef __attribute__((ext_vector_type(8)))  float    v8f;

#define F_DIM 128
#define K_CL  256

#define LOG2E 1.4426950408889634f
#define LN2   0.6931471805599453f

// ---------------------------------------------------------------------------
// Prep: per-cluster (||m_k||^2, -log2e/s1^2) pairs, inter-entropy (tiny), and
// pack m^T into the CDNA5 WMMA B-matrix (32x16 f16) lane layout:
//   flat = (((t*4+s)*32 + lane)*16 + i)
//   lane -> N = lane&15, K = (i&7) + (lane>>4)*8 + (i>>3)*16 within the 32-feature step s.
// ---------------------------------------------------------------------------
__global__ __launch_bounds__(256) void prep_kernel(
    const float* __restrict__ m, const float* __restrict__ s1,
    const float* __restrict__ s2, double* __restrict__ accum,
    float* __restrict__ inter_out, float2* __restrict__ cpar,
    _Float16* __restrict__ bpack)
{
    __shared__ float sh_xm[F_DIM];
    __shared__ float red[K_CL];
    const int tid = threadIdx.x;

    if (tid == 0) *accum = 0.0;   // must re-zero every launch (graph replay)

    if (tid < F_DIM) {
        float s = 0.f;
        for (int k = 0; k < K_CL; ++k) s += m[k * F_DIM + tid];
        sh_xm[tid] = s * (1.0f / (float)K_CL);
    }
    __syncthreads();

    float mn = 0.f, dd = 0.f;
    for (int f = 0; f < F_DIM; ++f) {
        float v = m[tid * F_DIM + f];
        mn += v * v;
        float df = sh_xm[f] - v;
        dd += df * df;
    }
    float s1v = s1[tid];
    cpar[tid] = make_float2(mn, -LOG2E / (s1v * s1v));
    float s2v = s2[tid];
    float g = -dd / (s2v * s2v);

    // softmax + entropy over the 256 clusters (inter term)
    red[tid] = g; __syncthreads();
    for (int off = 128; off > 0; off >>= 1) {
        if (tid < off) red[tid] = fmaxf(red[tid], red[tid + off]);
        __syncthreads();
    }
    float gmax = red[0]; __syncthreads();
    float e = __expf(g - gmax);
    red[tid] = e; __syncthreads();
    for (int off = 128; off > 0; off >>= 1) {
        if (tid < off) red[tid] += red[tid + off];
        __syncthreads();
    }
    float S = red[0]; __syncthreads();
    red[tid] = e * (g - gmax); __syncthreads();
    for (int off = 128; off > 0; off >>= 1) {
        if (tid < off) red[tid] += red[tid + off];
        __syncthreads();
    }
    float T = red[0];
    if (tid == 0) *inter_out = __logf(S) - T / S;

    // pack B into WMMA layout
    for (int fidx = tid; fidx < K_CL * F_DIM; fidx += blockDim.x) {
        int i = fidx & 15;
        int l = (fidx >> 4) & 31;
        int s = (fidx >> 9) & 3;
        int t = fidx >> 11;
        int cluster = t * 16 + (l & 15);
        int feat    = s * 32 + ((l >> 4) * 8) + (i & 7) + ((i >> 3) * 16);
        bpack[fidx] = (_Float16)m[cluster * F_DIM + feat];
    }
}

// ---------------------------------------------------------------------------
// Main fused kernel: per wave, 16 rows x 256 clusters via 64 WMMA f16 ops.
// GEMM runs on UNNORMALIZED x (f16); 1/||x|| folds into the logit epilogue.
// Logits are bounded in [-O(10), 0], so softmax needs no max subtraction.
// Entire softmax-entropy runs in base-2 (v_exp_f32/v_log_f32 native):
//   g2 = -d2 * log2e/s1^2,  S = sum 2^g2,  T2 = sum g2*2^g2,
//   H  = ln2 * (log2 S - T2/S).
// ---------------------------------------------------------------------------
__global__ __launch_bounds__(256, 1) void intra_kernel(
    const float* __restrict__ x, const float2* __restrict__ cpar,
    const _Float16* __restrict__ bpack, double* __restrict__ accum, int nTiles)
{
    __shared__ __align__(32) _Float16 sB[K_CL * F_DIM];  // 64 KB, reused by 8 waves
    __shared__ float2 sC[K_CL];                          // (mnorm2, -log2e/s1^2)

    const int tid = threadIdx.x;
    {
        const uint4* src = (const uint4*)bpack;
        uint4* dst = (uint4*)sB;
        for (int i = tid; i < (K_CL * F_DIM) / 8; i += blockDim.x) dst[i] = src[i];
    }
    sC[tid] = cpar[tid];
    __syncthreads();

    const int wave   = tid >> 5;
    const int lane   = tid & 31;
    const int laneM  = lane & 15;
    const int laneHi = lane >> 4;
    const int hiK    = laneHi * 8;
    const int stride = gridDim.x * 8;

    for (int tile = blockIdx.x * 8 + wave; tile < nTiles; tile += stride) {
        const float* rowp = x + (size_t)(tile * 16 + laneM) * F_DIM;

        // Load + immediately convert this lane's half-row (64 features) into
        // the WMMA A layout; accumulate sum of squares on the fly.
        v16h A[4];
        float xx = 0.f;
        #pragma unroll
        for (int s = 0; s < 4; ++s) {
            int base = 32 * s + hiK;
            float4 a0 = *(const float4*)(rowp + base);
            float4 a1 = *(const float4*)(rowp + base + 4);
            float4 a2 = *(const float4*)(rowp + base + 16);
            float4 a3 = *(const float4*)(rowp + base + 20);
            xx += a0.x*a0.x + a0.y*a0.y + a0.z*a0.z + a0.w*a0.w;
            xx += a1.x*a1.x + a1.y*a1.y + a1.z*a1.z + a1.w*a1.w;
            xx += a2.x*a2.x + a2.y*a2.y + a2.z*a2.z + a2.w*a2.w;
            xx += a3.x*a3.x + a3.y*a3.y + a3.z*a3.z + a3.w*a3.w;
            A[s][0]  = (_Float16)a0.x; A[s][1]  = (_Float16)a0.y;
            A[s][2]  = (_Float16)a0.z; A[s][3]  = (_Float16)a0.w;
            A[s][4]  = (_Float16)a1.x; A[s][5]  = (_Float16)a1.y;
            A[s][6]  = (_Float16)a1.z; A[s][7]  = (_Float16)a1.w;
            A[s][8]  = (_Float16)a2.x; A[s][9]  = (_Float16)a2.y;
            A[s][10] = (_Float16)a2.z; A[s][11] = (_Float16)a2.w;
            A[s][12] = (_Float16)a3.x; A[s][13] = (_Float16)a3.y;
            A[s][14] = (_Float16)a3.z; A[s][15] = (_Float16)a3.w;
        }
        xx += __shfl_xor(xx, 16, 32);          // lane pair covers the full row

        // Pull the x-stream ahead for the next grid-stride iteration
        // (global_prefetch_b8; speculative, dropped if OOB).
        {
            int nextTile = tile + stride;
            if (nextTile < nTiles)
                __builtin_prefetch(x + (size_t)(nextTile * 16 + laneM) * F_DIM + hiK, 0, 1);
        }

        // Per accumulator slot r (row laneHi*8+r): broadcast that row's xx,
        // then derive -2/||x|| and ||xn||^2 locally (v_rsq_f32, no divides).
        float n2i[8], xr[8];
        #pragma unroll
        for (int r = 0; r < 8; ++r) {
            float xxr = __shfl(xx, hiK + r, 32);
            float ivr = __builtin_amdgcn_rsqf(fmaxf(xxr, 1e-24f));
            n2i[r] = -2.0f * ivr;
            xr[r]  = xxr * ivr * ivr;
        }

        float Ss[8], Ts[8];
        #pragma unroll
        for (int r = 0; r < 8; ++r) { Ss[r] = 0.f; Ts[r] = 0.f; }

        #pragma unroll 4
        for (int t = 0; t < 16; ++t) {
            // Hoist the per-cluster constants: LDS latency hides under WMMA.
            float2 cp = sC[t * 16 + laneM];       // (mnorm2, -log2e/s1^2)

            v8f acc = {};
            const v16h* Bp = ((const v16h*)sB) + (t * 4) * 32 + lane;
            #pragma unroll
            for (int s = 0; s < 4; ++s) {
                v16h b = Bp[s * 32];
                acc = __builtin_amdgcn_wmma_f32_16x16x32_f16(
                    false, A[s], false, b, (short)0, acc, false, false);
            }

            // Split phases so the 8 exp chains are independent (hides TRANS latency).
            float g2v[8], ev[8];
            #pragma unroll
            for (int r = 0; r < 8; ++r) {
                float d2 = fmaxf(fmaf(n2i[r], acc[r], xr[r] + cp.x), 0.f);
                g2v[r] = d2 * cp.y;               // base-2 logit, in [-O(14), 0]
            }
            #pragma unroll
            for (int r = 0; r < 8; ++r) ev[r] = __builtin_amdgcn_exp2f(g2v[r]);
            #pragma unroll
            for (int r = 0; r < 8; ++r) {
                Ss[r] += ev[r];
                Ts[r]  = fmaf(g2v[r], ev[r], Ts[r]);
            }
        }

        // sum partial S/T across the 16 lanes sharing each row
        #pragma unroll
        for (int off = 1; off < 16; off <<= 1) {
            #pragma unroll
            for (int r = 0; r < 8; ++r) {
                Ss[r] += __shfl_xor(Ss[r], off, 32);
                Ts[r] += __shfl_xor(Ts[r], off, 32);
            }
        }

        // H_row = ln2 * (log2 S - T2/S)
        float h = 0.f;
        #pragma unroll
        for (int r = 0; r < 8; ++r)
            h += __builtin_amdgcn_logf(Ss[r]) - Ts[r] * __builtin_amdgcn_rcpf(Ss[r]);
        h *= LN2;

        float tot = (laneM == 0) ? h : 0.f;   // lanes 0 (rows 0-7) and 16 (rows 8-15)
        tot += __shfl_xor(tot, 16, 32);
        if (lane == 0) unsafeAtomicAdd(accum, (double)tot);
    }
}

__global__ void finalize_kernel(const double* __restrict__ accum,
                                const float* __restrict__ inter_ws,
                                float* __restrict__ out, int N)
{
    float intra = (float)(*accum / (double)N);
    float inter = *inter_ws;
    out[0] = intra - inter;
    out[1] = intra;
    out[2] = inter;
}

extern "C" void kernel_launch(void* const* d_in, const int* in_sizes, int n_in,
                              void* d_out, int out_size, void* d_ws, size_t ws_size,
                              hipStream_t stream)
{
    const float* x  = (const float*)d_in[0];
    const float* m  = (const float*)d_in[1];
    const float* s1 = (const float*)d_in[2];
    const float* s2 = (const float*)d_in[3];
    float* out = (float*)d_out;

    const int N = in_sizes[0] / F_DIM;

    char* ws = (char*)d_ws;
    double*   accum = (double*)ws;
    float*    inter = (float*)(ws + 8);
    float2*   cpar  = (float2*)(ws + 16);
    _Float16* bpack = (_Float16*)(ws + 16 + 2048);

    prep_kernel<<<1, 256, 0, stream>>>(m, s1, s2, accum, inter, cpar, bpack);

    const int nTiles = N / 16;
    intra_kernel<<<512, 256, 0, stream>>>(x, cpar, bpack, accum, nTiles);

    finalize_kernel<<<1, 1, 0, stream>>>(accum, inter, out, N);
}
